// GeneralizedCrossMessage_55301998903834
// MI455X (gfx1250) — matrix operand order repaired and checked by reading, not compiled, not run on hardware
//
#include <hip/hip_runtime.h>
#include <hip/hip_fp16.h>

typedef __attribute__((ext_vector_type(16))) _Float16 v16h;
typedef __attribute__((ext_vector_type(8)))  float    v8f;

#define BDIM   2
#define SDIM   1024
#define HDIM   768
#define DDIM   24
#define FD     96      // 4*D pair-feature width
#define HID    96      // score MLP hidden
#define OUTHID 768     // val MLP hidden
#define VALIN  2304    // 3*H
#define PKW    64      // padded pairwise-K (real K = 48)

// ---------------------------------------------------------------------------
// Low-dim projections: Zj = H_j @ W_pj^T, Zi = H_i @ W_pi^T  (tiny, VALU)
// ---------------------------------------------------------------------------
__global__ void proj_kernel(const float* __restrict__ Hj, const float* __restrict__ Hi,
                            const float* __restrict__ Wpj, const float* __restrict__ Wpi,
                            float* __restrict__ Zj, float* __restrict__ Zi)
{
    const int total = BDIM * SDIM * DDIM;
    int idx = blockIdx.x * blockDim.x + threadIdx.x;
    if (idx >= 2 * total) return;
    int sel = (idx >= total);
    int r = sel ? idx - total : idx;
    int token = r / DDIM, d = r % DDIM;
    const float* hrow = (sel ? Hi : Hj) + (size_t)token * HDIM;
    const float* wrow = (sel ? Wpi : Wpj) + (size_t)d * HDIM;
    float acc = 0.f;
#pragma unroll 8
    for (int h = 0; h < HDIM; ++h) acc += hrow[h] * wrow[h];
    (sel ? Zi : Zj)[r] = acc;
}

// ---------------------------------------------------------------------------
// Per-token halves of the score-MLP first layer:
//   Aj[t][n] = b1[n] + sum_d W_s1[n][d]    * Zj[t][d]
//   Bi[t][n] =         sum_d W_s1[n][24+d] * Zi[t][d]
// ---------------------------------------------------------------------------
__global__ void pairpre_kernel(const float* __restrict__ Zj, const float* __restrict__ Zi,
                               const float* __restrict__ W_s1, const float* __restrict__ b_s1,
                               float* __restrict__ Aj, float* __restrict__ Bi)
{
    const int total = BDIM * SDIM * HID;
    int idx = blockIdx.x * blockDim.x + threadIdx.x;
    if (idx >= 2 * total) return;
    int sel = (idx >= total);
    int r = sel ? idx - total : idx;
    int token = r / HID, n = r % HID;
    const float* z = (sel ? Zi : Zj) + (size_t)token * DDIM;
    const float* w = W_s1 + (size_t)n * FD + (sel ? DDIM : 0);
    float acc = sel ? 0.f : b_s1[n];
#pragma unroll
    for (int d = 0; d < DDIM; ++d) acc += z[d] * w[d];
    (sel ? Bi : Aj)[r] = acc;
}

// ---------------------------------------------------------------------------
// f32 -> f16 conversion (plain)
// ---------------------------------------------------------------------------
__global__ void cvt_f16(const float* __restrict__ src, _Float16* __restrict__ dst, size_t n)
{
    size_t i = (size_t)blockIdx.x * blockDim.x + threadIdx.x;
    if (i < n) dst[i] = (_Float16)src[i];
}

// f32 [N][K] -> f16 [K][N] transpose (one-time weight re-layout)
__global__ void cvt_f16_t(const float* __restrict__ src, _Float16* __restrict__ dst,
                          int N, int K)
{
    size_t i = (size_t)blockIdx.x * blockDim.x + threadIdx.x;
    size_t total = (size_t)N * K;
    if (i >= total) return;
    int k = (int)(i / N), n = (int)(i % N);
    dst[i] = (_Float16)src[(size_t)n * K + k];
}

// ---------------------------------------------------------------------------
// msg_in = [ctx, H_j, ctx*H_j] as f16
// ---------------------------------------------------------------------------
__global__ void build_msgin(const float* __restrict__ ctx, const float* __restrict__ Hj,
                            _Float16* __restrict__ dst)
{
    size_t i = (size_t)blockIdx.x * blockDim.x + threadIdx.x;
    const size_t total = (size_t)BDIM * SDIM * 3 * HDIM;
    if (i >= total) return;
    size_t row = i / (3 * HDIM);
    int c = (int)(i % (3 * HDIM));
    float v;
    if (c < HDIM)            v = ctx[row * HDIM + c];
    else if (c < 2 * HDIM)   v = Hj[row * HDIM + (c - HDIM)];
    else { int cc = c - 2 * HDIM; v = ctx[row * HDIM + cc] * Hj[row * HDIM + cc]; }
    dst[i] = (_Float16)v;
}

// ---------------------------------------------------------------------------
// Pair score MLP (pairwise half only, K=48 padded to 64, 2 WMMA K-steps).
// One block = 16 j's x 64 i's. A (256 pairs x 48 feats) built in registers
// from Zj/Zi LDS tiles; per-token halves Aj/Bi added in the epilogue,
// then relu * W_s2 reduced in-register -> logits.
// ---------------------------------------------------------------------------
__global__ __launch_bounds__(256) void score_kernel(
    const float* __restrict__ Zj, const float* __restrict__ Zi,
    const _Float16* __restrict__ W1h, const float* __restrict__ AjG,
    const float* __restrict__ BiG, const float* __restrict__ w2,
    const float* __restrict__ b2, float* __restrict__ logits)
{
    __shared__ _Float16 sW1p[HID][PKW + 8];   // [n][k] pairwise cols of W_s1 (zero-padded)
    __shared__ float    sW2[HID];
    __shared__ float    sAj[16][HID + 4];     // per-j first-layer partial (incl. b1)
    __shared__ float    sBi[16][HID + 4];     // per-i first-layer partial
    __shared__ float    sZj[16][DDIM + 1];
    __shared__ float    sZi[16][DDIM + 1];
    __shared__ float    sLogit[16][16];       // [jj][ii]

    const int tid  = threadIdx.x;
    const int b    = blockIdx.y;
    const int jt   = blockIdx.x >> 4;         // 64 j-tiles
    const int ic   = blockIdx.x & 15;         // 16 i-chunks of 64
    const int j0   = jt * 16;
    const int wave = tid >> 5, lane = tid & 31;
    const int hs   = lane >> 4, lr = lane & 15;

    // pairwise columns of W_s1: feature k<24 -> hadamard col (48+k), k<48 -> absdiff
    for (int idx = tid; idx < HID * PKW; idx += 256) {
        int n = idx / PKW, k = idx % PKW;
        sW1p[n][k] = (k < 48) ? W1h[n * FD + 48 + k] : (_Float16)0.f;
    }
    for (int idx = tid; idx < HID; idx += 256) sW2[idx] = w2[idx];
    for (int idx = tid; idx < 16 * HID; idx += 256) {
        int r = idx / HID, n = idx % HID;
        sAj[r][n] = AjG[((size_t)b * SDIM + j0 + r) * HID + n];
    }
    if (tid < 16 * DDIM) {
        int r = tid / DDIM, d = tid % DDIM;
        sZj[r][d] = Zj[((size_t)b * SDIM + j0 + r) * DDIM + d];
    }
    const float bs2 = b2[0];

    for (int sub = 0; sub < 4; ++sub) {
        const int i0 = ic * 64 + sub * 16;
        __syncthreads();
        if (tid < 16 * DDIM) {
            int r = tid / DDIM, d = tid % DDIM;
            sZi[r][d] = Zi[((size_t)b * SDIM + i0 + r) * DDIM + d];
        }
        for (int idx = tid; idx < 16 * HID; idx += 256) {
            int r = idx / HID, n = idx % HID;
            sBi[r][n] = BiG[((size_t)b * SDIM + i0 + r) * HID + n];
        }
        __syncthreads();

        for (int mi = 0; mi < 2; ++mi) {
            const int m = wave * 2 + mi;      // jj this wave owns (rows = all 16 ii)
            // A fragments (ISA wave32 A layout): lane row = ii = lr
            v16h afrag[2];
#pragma unroll
            for (int ks = 0; ks < 2; ++ks) {
#pragma unroll
                for (int e = 0; e < 16; ++e) {
                    int k = ks * 32 + hs * 8 + (e & 7) + ((e >> 3) << 4);
                    float f;
                    if (k < 24)       f = sZj[m][k] * sZi[lr][k];
                    else if (k < 48) { int d = k - 24; f = fabsf(sZj[m][d] - sZi[lr][d]); }
                    else              f = 0.f;
                    afrag[ks][e] = (_Float16)f;
                }
            }
            float vsum[8];
#pragma unroll
            for (int r8 = 0; r8 < 8; ++r8) vsum[r8] = 0.f;

            for (int nt = 0; nt < 6; ++nt) {
                const int n = nt * 16 + lr;
                v8f c = {0.f, 0.f, 0.f, 0.f, 0.f, 0.f, 0.f, 0.f};
#pragma unroll
                for (int ks = 0; ks < 2; ++ks) {
                    v16h bfrag;
                    const int kb = ks * 32;
#pragma unroll
                    for (int e = 0; e < 16; ++e) bfrag[e] = sW1p[n][kb + hs * 8 + (e & 7) + ((e >> 3) << 4)];
                    c = __builtin_amdgcn_wmma_f32_16x16x32_f16(
                            false, afrag[ks], false, bfrag, (short)0, c, false, false);
                }
                const float w2n = sW2[n];
                const float ajn = sAj[m][n];
#pragma unroll
                for (int r8 = 0; r8 < 8; ++r8) {
                    const int row = r8 + 8 * hs;  // ii
                    float v = c[r8] + ajn + sBi[row][n];
                    v = v > 0.f ? v : 0.f;
                    vsum[r8] += v * w2n;
                }
            }
            // reduce over the 16 columns (16-lane groups), store logit rows
#pragma unroll
            for (int r8 = 0; r8 < 8; ++r8) {
                float v = vsum[r8];
#pragma unroll
                for (int off = 1; off < 16; off <<= 1) v += __shfl_xor(v, off, 32);
                if (lr == 0) sLogit[m][r8 + 8 * hs] = v;
            }
        }
        __syncthreads();
        {
            const int jj = tid >> 4, ii = tid & 15;
            logits[((size_t)b * SDIM + (j0 + jj)) * SDIM + (i0 + ii)] = sLogit[jj][ii] + bs2;
        }
    }
}

// ---------------------------------------------------------------------------
// Row softmax with additive key mask; writes f16 probs.
// ---------------------------------------------------------------------------
__global__ __launch_bounds__(256) void softmax_kernel(
    const float* __restrict__ logits, const float* __restrict__ mask,
    _Float16* __restrict__ P)
{
    __shared__ float red[256];
    const int tid = threadIdx.x;
    const int b = blockIdx.x / SDIM;
    const int j = blockIdx.x % SDIM;
    const size_t rowoff = ((size_t)b * SDIM + j) * SDIM;
    const float neg = -3.402823466e38f;
    float l[4];
    float mx = neg;
#pragma unroll
    for (int t = 0; t < 4; ++t) {
        int i = t * 256 + tid;
        float v = logits[rowoff + i] + (1.0f - mask[(size_t)b * SDIM + i]) * neg;
        l[t] = v; mx = fmaxf(mx, v);
    }
    red[tid] = mx; __syncthreads();
    for (int s = 128; s > 0; s >>= 1) {
        if (tid < s) red[tid] = fmaxf(red[tid], red[tid + s]);
        __syncthreads();
    }
    mx = red[0]; __syncthreads();
    float sum = 0.f;
#pragma unroll
    for (int t = 0; t < 4; ++t) { float e = __expf(l[t] - mx); l[t] = e; sum += e; }
    red[tid] = sum; __syncthreads();
    for (int s = 128; s > 0; s >>= 1) {
        if (tid < s) red[tid] += red[tid + s];
        __syncthreads();
    }
    const float inv = 1.0f / red[0];
#pragma unroll
    for (int t = 0; t < 4; ++t) P[rowoff + t * 256 + tid] = (_Float16)(l[t] * inv);
}

// ---------------------------------------------------------------------------
// Generic f16 WMMA GEMM with CDNA5 async global->LDS tile staging.
//   out[m,n] = act( sum_k A[m,k]*B[k][n] + bias[n] ) * alpha
//   A: f16 [M][K] row-major (M from grid.y*64), B: f16 [K][N] row-major.
//   Tiles staged with global_load_async_to_lds_b128 (ASYNCcnt), 512B/wave/op.
// ---------------------------------------------------------------------------
__global__ __launch_bounds__(256) void gemm_wmma(
    const _Float16* __restrict__ A, const _Float16* __restrict__ Bm,
    const float* __restrict__ bias, const float* __restrict__ alphaPtr,
    _Float16* __restrict__ outH, float* __restrict__ outF,
    int N, int K,
    long long sAb, long long sBb, long long sOb,
    int relu)
{
    // sA stride 40 halves = 80B: 16B-aligned rows, and 20*r mod 64 hits 16
    // distinct banks for r=0..15 -> conflict-free A-fragment reads.
    __shared__ __align__(16) _Float16 sA[64][40];
    __shared__ __align__(16) _Float16 sB[32][64];

    const int tid  = threadIdx.x;
    const int wave = tid >> 5, lane = tid & 31;
    const int hs   = lane >> 4, lr = lane & 15;
    const int mt   = wave & 3;
    const int nt0  = (wave >> 2) * 2;
    const int m0   = blockIdx.y * 64, n0 = blockIdx.x * 64;
    const long long zb = blockIdx.z;
    const _Float16* Ab = A + zb * sAb;
    const _Float16* Bb = Bm + zb * sBb;

    // per-lane async-copy slots: A tile 64 rows x 4 x 16B, B tile 32 rows x 8 x 16B
    const int ar = tid >> 2, ac = (tid & 3) << 3;   // halves
    const int bk = tid >> 3, bc = (tid & 7) << 3;   // halves
    const unsigned aL = (unsigned)(uintptr_t)&sA[ar][ac];
    const unsigned bL = (unsigned)(uintptr_t)&sB[bk][bc];

    v8f acc0 = {0.f,0.f,0.f,0.f,0.f,0.f,0.f,0.f};
    v8f acc1 = acc0;

    for (int ks = 0; ks < K; ks += 32) {
        __syncthreads();
        const unsigned aG = (unsigned)((((size_t)(m0 + ar)) * K + ks + ac) * 2);
        const unsigned bG = (unsigned)((((size_t)(ks + bk)) * N + n0 + bc) * 2);
        asm volatile("global_load_async_to_lds_b128 %0, %1, %2"
                     :: "v"(aL), "v"(aG), "s"(Ab) : "memory");
        asm volatile("global_load_async_to_lds_b128 %0, %1, %2"
                     :: "v"(bL), "v"(bG), "s"(Bb) : "memory");
        asm volatile("s_wait_asynccnt 0x0" ::: "memory");
        __syncthreads();

        v16h af, bf0, bf1;
#pragma unroll
        for (int e = 0; e < 16; ++e) {
            int kk = hs * 8 + (e & 7) + ((e >> 3) << 4);
            af[e] = sA[mt * 16 + lr][kk];
        }
#pragma unroll
        for (int e = 0; e < 16; ++e) {
            int kk = hs * 8 + (e & 7) + ((e >> 3) << 4);
            bf0[e] = sB[kk][nt0 * 16 + lr];
            bf1[e] = sB[kk][(nt0 + 1) * 16 + lr];
        }
        acc0 = __builtin_amdgcn_wmma_f32_16x16x32_f16(false, af, false, bf0, (short)0, acc0, false, false);
        acc1 = __builtin_amdgcn_wmma_f32_16x16x32_f16(false, af, false, bf1, (short)0, acc1, false, false);
    }

    const float alpha = alphaPtr ? alphaPtr[0] : 1.0f;
#pragma unroll
    for (int t = 0; t < 2; ++t) {
        v8f acc = t ? acc1 : acc0;
        const int n = n0 + (nt0 + t) * 16 + lr;
        const float bv = bias ? bias[n] : 0.f;
#pragma unroll
        for (int r8 = 0; r8 < 8; ++r8) {
            const int m = m0 + mt * 16 + r8 + 8 * hs;
            float v = acc[r8] + bv;
            if (relu) v = v > 0.f ? v : 0.f;
            v *= alpha;
            size_t oi = (size_t)(zb * sOb) + (size_t)m * N + n;
            if (outF) outF[oi] = v;
            else      outH[oi] = (_Float16)v;
        }
    }
}

// ---------------------------------------------------------------------------
extern "C" void kernel_launch(void* const* d_in, const int* in_sizes, int n_in,
                              void* d_out, int out_size, void* d_ws, size_t ws_size,
                              hipStream_t stream)
{
    (void)in_sizes; (void)n_in; (void)out_size; (void)ws_size;
    const float* H_j  = (const float*)d_in[0];
    const float* H_i  = (const float*)d_in[1];
    const float* mask = (const float*)d_in[2];
    const float* W_pj = (const float*)d_in[3];
    const float* W_pi = (const float*)d_in[4];
    const float* W_s1 = (const float*)d_in[5];
    const float* b_s1 = (const float*)d_in[6];
    const float* W_s2 = (const float*)d_in[7];
    const float* b_s2 = (const float*)d_in[8];
    const float* W_v1 = (const float*)d_in[9];
    const float* b_v1 = (const float*)d_in[10];
    const float* W_v2 = (const float*)d_in[11];
    const float* b_v2 = (const float*)d_in[12];
    const float* alpha = (const float*)d_in[13];

    char* ws = (char*)d_ws;
    size_t off = 0;
    auto take = [&](size_t bytes) -> char* {
        char* p = ws + off;
        off += (bytes + 255) & ~(size_t)255;
        return p;
    };

    float*     zj     = (float*)    take((size_t)BDIM * SDIM * DDIM * 4);
    float*     zi     = (float*)    take((size_t)BDIM * SDIM * DDIM * 4);
    float*     aj     = (float*)    take((size_t)BDIM * SDIM * HID * 4);
    float*     bi     = (float*)    take((size_t)BDIM * SDIM * HID * 4);
    _Float16*  w1h    = (_Float16*) take((size_t)HID * FD * 2);
    _Float16*  wv1t   = (_Float16*) take((size_t)VALIN * OUTHID * 2);   // [K][N]
    _Float16*  wv2t   = (_Float16*) take((size_t)OUTHID * HDIM * 2);    // [K][N]
    _Float16*  hih    = (_Float16*) take((size_t)BDIM * SDIM * HDIM * 2);
    float*     logits = (float*)    take((size_t)BDIM * SDIM * SDIM * 4);
    _Float16*  ph     = (_Float16*) take((size_t)BDIM * SDIM * SDIM * 2);
    float*     ctx    = (float*)    take((size_t)BDIM * SDIM * HDIM * 4);
    _Float16*  msgin  = (_Float16*) take((size_t)BDIM * SDIM * VALIN * 2);
    _Float16*  hid    = (_Float16*) take((size_t)BDIM * SDIM * OUTHID * 2);

    // 1) projections
    {
        int n = 2 * BDIM * SDIM * DDIM;
        proj_kernel<<<(n + 255) / 256, 256, 0, stream>>>(H_j, H_i, W_pj, W_pi, zj, zi);
    }
    // 2) per-token score-MLP halves
    {
        int n = 2 * BDIM * SDIM * HID;
        pairpre_kernel<<<(n + 255) / 256, 256, 0, stream>>>(zj, zi, W_s1, b_s1, aj, bi);
    }
    // 3) weight / input conversions (val weights pre-transposed to [K][N])
    {
        size_t n;
        n = (size_t)HID * FD;
        cvt_f16<<<(unsigned)((n + 255) / 256), 256, 0, stream>>>(W_s1, w1h, n);
        n = (size_t)OUTHID * VALIN;
        cvt_f16_t<<<(unsigned)((n + 255) / 256), 256, 0, stream>>>(W_v1, wv1t, OUTHID, VALIN);
        n = (size_t)HDIM * OUTHID;
        cvt_f16_t<<<(unsigned)((n + 255) / 256), 256, 0, stream>>>(W_v2, wv2t, HDIM, OUTHID);
        n = (size_t)BDIM * SDIM * HDIM;
        cvt_f16<<<(unsigned)((n + 255) / 256), 256, 0, stream>>>(H_i, hih, n);
    }
    // 4) pairwise score MLP -> logits
    score_kernel<<<dim3(1024, BDIM), 256, 0, stream>>>(zj, zi, w1h, aj, bi, W_s2, b_s2, logits);
    // 5) softmax -> f16 probs
    softmax_kernel<<<BDIM * SDIM, 256, 0, stream>>>(logits, mask, ph);
    // 6) ctx = P @ H_i (batched)
    gemm_wmma<<<dim3(HDIM / 64, SDIM / 64, BDIM), 256, 0, stream>>>(
        ph, hih, nullptr, nullptr, nullptr, ctx,
        HDIM, SDIM,
        (long long)SDIM * SDIM, (long long)SDIM * HDIM, (long long)SDIM * HDIM,
        /*relu=*/0);
    // 7) msg_in = [ctx, H_j, ctx*H_j]
    {
        size_t n = (size_t)BDIM * SDIM * VALIN;
        build_msgin<<<(unsigned)((n + 255) / 256), 256, 0, stream>>>(ctx, H_j, msgin);
    }
    // 8) hid = relu(msg_in @ W_v1^T + b_v1)
    gemm_wmma<<<dim3(OUTHID / 64, (BDIM * SDIM) / 64, 1), 256, 0, stream>>>(
        msgin, wv1t, b_v1, nullptr, hid, nullptr,
        OUTHID, VALIN, 0, 0, 0, /*relu=*/1);
    // 9) out = alpha * (hid @ W_v2^T + b_v2)
    gemm_wmma<<<dim3(HDIM / 64, (BDIM * SDIM) / 64, 1), 256, 0, stream>>>(
        hid, wv2t, b_v2, alpha, nullptr, (float*)d_out,
        HDIM, OUTHID, 0, 0, 0, /*relu=*/0);
}
